// GraphAttentionLayer_10230612099213
// MI455X (gfx1250) — matrix operand order repaired and checked by reading, not compiled, not run on hardware
//
#include <hip/hip_runtime.h>

// GAT layer for MI455X (gfx1250, wave32).
// Memory-bound on the 134MB adjacency -> single streaming pass, fused
// exp/softmax-normalize/aggregate with v_wmma_f32_16x16x32_bf16.

#define B_ 8
#define N_ 2048
#define F_ 256
#define D_ 64

typedef __attribute__((ext_vector_type(16))) __bf16 v16bf;
typedef __attribute__((ext_vector_type(8)))  __bf16 v8bf;
typedef __attribute__((ext_vector_type(8)))  float  v8f;

__device__ __forceinline__ v8f vzero8() {
    v8f z;
#pragma unroll
    for (int i = 0; i < 8; ++i) z[i] = 0.f;
    return z;
}

// ---------------------------------------------------------------------------
// k0: pre-swizzle W (256x64 f32, row-major) into bf16 WMMA B-matrix layout.
// WB element index e = ((kc*4 + t)*32 + lane)*16 + i
//   K = kc*32 + 8*(lane>>4) + (i&7) + 16*(i>>3)   (table order: i<8 -> K 8h+i, i>=8 -> K 16+8h+i-8)
//   d = 16*t + (lane&15)
// ---------------------------------------------------------------------------
__global__ __launch_bounds__(256) void k0_swizzle_w(const float* __restrict__ W,
                                                    __bf16* __restrict__ WB) {
    const int total = 8 * 4 * 32 * 16; // 16384
    for (int e = blockIdx.x * 256 + threadIdx.x; e < total; e += gridDim.x * 256) {
        int i    = e & 15;
        int lane = (e >> 4) & 31;
        int t    = (e >> 9) & 3;
        int kc   = e >> 11;
        int K = kc * 32 + 8 * (lane >> 4) + (i & 7) + 16 * (i >> 3);
        int d = t * 16 + (lane & 15);
        WB[e] = (__bf16)W[K * D_ + d];
    }
}

// ---------------------------------------------------------------------------
// k1: h = relu(X @ W) via WMMA bf16; store hT[b][d][j] (bf16) for pass-2 B
// tiles; reduce si = h.a2, sj = h.a1 with wave shuffles.
// One wave handles 16 flat rows of X (B*N/16 = 1024 tiles, 8 waves/block).
// ---------------------------------------------------------------------------
__global__ __launch_bounds__(256) void k1_gemm_h(const float* __restrict__ X,
                                                 const __bf16* __restrict__ WB,
                                                 const float* __restrict__ attk,
                                                 __bf16* __restrict__ hT,
                                                 float* __restrict__ si,
                                                 float* __restrict__ sj) {
    const int lane  = threadIdx.x & 31;
    const int wave  = threadIdx.x >> 5;
    const int tile  = blockIdx.x * 8 + wave;   // 0..1023
    const int rbase = tile * 16;               // flat row over B*N
    const int half  = lane >> 4;
    const int ln    = lane & 15;

    v8f acc[4];
#pragma unroll
    for (int t = 0; t < 4; ++t) acc[t] = vzero8();

    // A rows: lane l (both halves) owns row M = ln
    const float* xrow = X + (size_t)(rbase + ln) * F_;

    for (int kb = 0; kb < F_; kb += 32) {
        // 16-bit A 16x32 layout: VGPR0-3 = K koff..koff+7, VGPR4-7 = K 16+koff..16+koff+7
        const int koff = kb + half * 8;
        v8f x0 = *(const v8f*)(xrow + koff);
        v8f x1 = *(const v8f*)(xrow + koff + 16);
        v16bf a;
#pragma unroll
        for (int c = 0; c < 8; ++c) {
            a[c]     = (__bf16)x0[c];
            a[8 + c] = (__bf16)x1[c];
        }
        const int kc = kb >> 5;
#pragma unroll
        for (int t = 0; t < 4; ++t) {
            v16bf bt = *(const v16bf*)(WB + ((size_t)((kc * 4 + t) * 32 + lane)) * 16);
            acc[t] = __builtin_amdgcn_wmma_f32_16x16x32_bf16(
                false, a, false, bt, (short)0, acc[t], false, false);
        }
    }

    // fused relu
#pragma unroll
    for (int t = 0; t < 4; ++t)
#pragma unroll
        for (int v = 0; v < 8; ++v) acc[t][v] = fmaxf(acc[t][v], 0.f);

    const int b   = rbase / N_;
    const int jb0 = rbase % N_;

    // store hT[b][d][j]: C/D layout -> lane owns d = 16t+ln, rows M = v + 8*half
#pragma unroll
    for (int t = 0; t < 4; ++t) {
        const int d = t * 16 + ln;
        v8bf hv;
#pragma unroll
        for (int v = 0; v < 8; ++v) hv[v] = (__bf16)acc[t][v];
        *(v8bf*)(hT + ((size_t)(b * D_ + d)) * N_ + jb0 + half * 8) = hv;
    }

    // si/sj: per-row dot with a2/a1, reduced over the 16 lanes of each half
    float a1v[4], a2v[4];
#pragma unroll
    for (int t = 0; t < 4; ++t) {
        a1v[t] = attk[t * 16 + ln];        // a1 = attention_kernel[:64]
        a2v[t] = attk[D_ + t * 16 + ln];   // a2 = attention_kernel[64:]
    }
#pragma unroll
    for (int v = 0; v < 8; ++v) {
        float p1 = 0.f, p2 = 0.f;
#pragma unroll
        for (int t = 0; t < 4; ++t) {
            p1 += acc[t][v] * a1v[t];
            p2 += acc[t][v] * a2v[t];
        }
#pragma unroll
        for (int m = 1; m < 16; m <<= 1) {
            p1 += __shfl_xor(p1, m);
            p2 += __shfl_xor(p2, m);
        }
        if (ln == 0) { // lane 0 stores rows 0..7, lane 16 stores rows 8..15
            sj[b * N_ + jb0 + half * 8 + v] = p1;
            si[b * N_ + jb0 + half * 8 + v] = p2;
        }
    }
}

// ---------------------------------------------------------------------------
// k2: fused attention + aggregation. One wave per 16-row tile of one batch.
// Streams adjacency once; p = exp(adj * relu(si+sj)); out = (P@h)/rowsum(P).
// ---------------------------------------------------------------------------
__global__ __launch_bounds__(256) void k2_attend(const float* __restrict__ adj,
                                                 const __bf16* __restrict__ hT,
                                                 const float* __restrict__ si,
                                                 const float* __restrict__ sj,
                                                 float* __restrict__ out) {
    const int lane  = threadIdx.x & 31;
    const int wave  = threadIdx.x >> 5;
    const int tile  = blockIdx.x * 8 + wave;     // 0..1023
    const int b     = tile / (N_ / 16);
    const int ibase = (tile % (N_ / 16)) * 16;
    const int half  = lane >> 4;
    const int ln    = lane & 15;

    const int    i      = ibase + ln;
    const float  s_i    = si[b * N_ + i];
    const float* adjrow = adj + ((size_t)b * N_ + i) * N_;
    const float* sjb    = sj + b * N_;
    const __bf16* hTb   = hT + (size_t)b * D_ * N_;

    v8f acc[4];
#pragma unroll
    for (int t = 0; t < 4; ++t) acc[t] = vzero8();
    float den = 0.f;

    for (int jb = 0; jb < N_; jb += 32) {
        if (jb + 256 < N_) __builtin_prefetch(adjrow + jb + 256, 0, 0); // global_prefetch_b8

        const int o0 = jb + half * 16;       // A K-slots 0..7 (K 16h..16h+7)
        const int o1 = jb + half * 16 + 8;   // A K-slots 8..15 (K 16h+8..16h+15)
        v8f ad0 = *(const v8f*)(adjrow + o0);
        v8f ad1 = *(const v8f*)(adjrow + o1);
        v8f sv0 = *(const v8f*)(sjb + o0);
        v8f sv1 = *(const v8f*)(sjb + o1);

        v16bf a;
#pragma unroll
        for (int c = 0; c < 8; ++c) {
            float p0 = __expf(ad0[c] * fmaxf(s_i + sv0[c], 0.f));
            float p1 = __expf(ad1[c] * fmaxf(s_i + sv1[c], 0.f));
            den += p0 + p1;
            a[c]     = (__bf16)p0;
            a[8 + c] = (__bf16)p1;
        }
#pragma unroll
        for (int t = 0; t < 4; ++t) {
            // B layout: lanes 0-15 K 0..15 of chunk, lanes 16-31 K 16..31; col = ln
            v16bf bt = *(const v16bf*)(hTb + (size_t)(t * 16 + ln) * N_ + jb + half * 16);
            acc[t] = __builtin_amdgcn_wmma_f32_16x16x32_bf16(
                false, a, false, bt, (short)0, acc[t], false, false);
        }
    }

    // full row denominator: halves cover complementary K sets
    den += __shfl_xor(den, 16);
    float dinv[8];
#pragma unroll
    for (int v = 0; v < 8; ++v) dinv[v] = 1.f / __shfl(den, v + half * 8);

    float* outb = out + ((size_t)b * N_ + ibase) * D_;
#pragma unroll
    for (int t = 0; t < 4; ++t) {
        const int d = t * 16 + ln;
#pragma unroll
        for (int v = 0; v < 8; ++v) {
            const int M = v + half * 8;
            outb[(size_t)M * D_ + d] = acc[t][v] * dinv[v];
        }
    }
}

// ---------------------------------------------------------------------------
extern "C" void kernel_launch(void* const* d_in, const int* in_sizes, int n_in,
                              void* d_out, int out_size, void* d_ws, size_t ws_size,
                              hipStream_t stream) {
    const float* adjacency = (const float*)d_in[0]; // (B,N,N)
    const float* node_feat = (const float*)d_in[1]; // (B,N,F)
    const float* kernelW   = (const float*)d_in[2]; // (F,D)
    const float* attk      = (const float*)d_in[3]; // (2D,1)
    float* out = (float*)d_out;                     // (B,N,D)

    char* ws = (char*)d_ws;
    __bf16* WB = (__bf16*)ws;                                   //  32 KB
    __bf16* hT = (__bf16*)(ws + 32768);                         //   2 MB
    float*  si = (float*)(ws + 32768 + 2097152);                //  64 KB
    float*  sj = (float*)(ws + 32768 + 2097152 + 65536);        //  64 KB

    k0_swizzle_w<<<64, 256, 0, stream>>>(kernelW, WB);
    k1_gemm_h<<<128, 256, 0, stream>>>(node_feat, WB, attk, hT, si, sj);
    k2_attend<<<128, 256, 0, stream>>>(adjacency, hT, si, sj, out);
}
